// Model_86045374808322
// MI455X (gfx1250) — matrix operand order repaired and checked by reading, not compiled, not run on hardware
//
#include <hip/hip_runtime.h>
#include <hip/hip_bf16.h>

typedef __attribute__((ext_vector_type(16))) _Float16 v16h;
typedef __attribute__((ext_vector_type(8)))  float    v8f;
typedef unsigned int v4u  __attribute__((ext_vector_type(4)));
typedef int          v8i_t __attribute__((ext_vector_type(8)));
typedef int          v4i_t __attribute__((ext_vector_type(4)));

constexpr int BN   = 128;   // batch
constexpr int CN   = 128;   // channels
constexpr int TN   = 2000;  // time
constexpr int TPAD = 2016;  // time padded to multiple of 32
constexpr int FN   = 64;    // TCN features
constexpr int HN   = 128;   // LIF hidden
constexpr int SEQN = 10;

#if defined(__has_builtin)
#if __has_builtin(__builtin_amdgcn_tensor_load_to_lds) && __has_builtin(__builtin_amdgcn_s_wait_tensorcnt)
#define HAVE_TDM 1
#endif
#endif

// ---------------- WMMA fragment helpers (16x16x32 f16 -> f32) ----------------
// A-matrix 16x32 (MxK), row-major source with leading dim ld (halves).
// Layout (ISA 7.12.2): lanes 0-15 hold row M=l, halves K {0..7, 16..23};
// lanes 16-31 hold K {8..15, 24..31}.
__device__ __forceinline__ v16h ld_a(const _Float16* p0, int ld, int lane) {
  int l = lane & 15;
  int hi = (lane >> 4) & 1;
  const _Float16* p = p0 + l * ld + hi * 8;
  v16h a;
#pragma unroll
  for (int i = 0; i < 8; ++i) { a[i] = p[i]; a[i + 8] = p[i + 16]; }
  return a;
}

// B-matrix 32x16 (KxN) given as BT (N rows x K cols, row-major, ld halves).
// Lane n<16 holds B[k0+{0..7,16..23}][n]; lane n+16 holds k0+{8..15,24..31}.
__device__ __forceinline__ v16h ld_bt(const _Float16* p0, int ld, int lane) {
  int l = lane & 15;
  int hi = (lane >> 4) & 1;
  const _Float16* p = p0 + l * ld + hi * 8;
  v16h b;
#pragma unroll
  for (int i = 0; i < 8; ++i) { b[i] = p[i]; b[i + 8] = p[i + 16]; }
  return b;
}

#define WMMA_F16(a, b, c) \
  __builtin_amdgcn_wmma_f32_16x16x32_f16(false, (a), false, (b), (short)0, (c), false, false)

// ---------------- K1: per-(b,c) mean over T + f16 padded copy of X ----------
__global__ void k_mean_xh(const float* __restrict__ X, float* __restrict__ mean,
                          _Float16* __restrict__ Xh) {
  int row = blockIdx.x;                       // b*CN + c
  const float* x = X + (size_t)row * TN;
  _Float16* xo = Xh + (size_t)row * TPAD;
  __shared__ float red[256];
  float s = 0.0f;
  for (int t = threadIdx.x; t < TN; t += 256) s += x[t];
  red[threadIdx.x] = s;
  __syncthreads();
  for (int o = 128; o > 0; o >>= 1) {
    if (threadIdx.x < o) red[threadIdx.x] += red[threadIdx.x + o];
    __syncthreads();
  }
  if (threadIdx.x == 0) mean[row] = red[0] * (1.0f / TN);
  for (int t = threadIdx.x; t < TPAD; t += 256)
    xo[t] = (t < TN) ? (_Float16)x[t] : (_Float16)0.0f;
}

// ---------------- K2: cov = (X Xt - T m mT)/(T-1) + 1e-4 I  (WMMA) ----------
// One wave per 16-row M-strip; A fragment loaded once per K-chunk and reused
// across all 8 N-tiles (8 live accumulators) to halve fragment traffic.
__global__ void k_cov(const _Float16* __restrict__ Xh, const float* __restrict__ mean,
                      float* __restrict__ cov) {
  int b = blockIdx.x;
  int wave = threadIdx.x >> 5, lane = threadIdx.x & 31;
  const _Float16* Xb = Xh + (size_t)b * CN * TPAD;
  const float* mb = mean + b * CN;
  float* covb = cov + (size_t)b * CN * CN;
  const float inv = 1.0f / (float)(TN - 1);
  const int m0 = wave * 16;
  v8f acc[8] = {{}, {}, {}, {}, {}, {}, {}, {}};
  for (int k0 = 0; k0 < TPAD; k0 += 32) {
    __builtin_prefetch(Xb + (size_t)m0 * TPAD + k0 + 64, 0, 1);
    v16h a = ld_a(Xb + (size_t)m0 * TPAD + k0, TPAD, lane);
#pragma unroll
    for (int nt = 0; nt < 8; ++nt) {
      v16h bb = ld_bt(Xb + (size_t)(nt * 16) * TPAD + k0, TPAD, lane);
      acc[nt] = WMMA_F16(a, bb, acc[nt]);
    }
  }
  const int n_lo = lane & 15;
  const int hi8 = ((lane >> 4) & 1) ? 8 : 0;
#pragma unroll
  for (int nt = 0; nt < 8; ++nt) {
    int n = nt * 16 + n_lo;
    float mn = mb[n];
#pragma unroll
    for (int r = 0; r < 8; ++r) {
      int mm = m0 + hi8 + r;
      float v = acc[nt][r] * inv - ((float)TN * inv) * mb[mm] * mn;
      if (mm == n) v += 1e-4f;
      covb[mm * CN + n] = v;
    }
  }
}

// ---- K3: batched Jacobi eigh (xor pairing) + logm + row softmax -> adj -----
__global__ void k_eigh_adj(const float* __restrict__ cov, float* __restrict__ adj) {
  extern __shared__ char dynsm[];
  float* A = (float*)dynsm;        // 128x128
  float* V = A + 16384;            // 128x128
  float* cs = V + 16384;           // 64 pairs * {c,s}
  float* logl = cs + 128;          // 128 eigenvalue logs
  const int b = blockIdx.x, tid = threadIdx.x;
  const float* Ain = cov + (size_t)b * 16384;
  float* Aout = adj + (size_t)b * 16384;
  for (int i = tid; i < 16384; i += 256) {
    A[i] = Ain[i];
    V[i] = ((i >> 7) == (i & 127)) ? 1.0f : 0.0f;
  }
  __syncthreads();
  for (int sweep = 0; sweep < 8; ++sweep) {
    for (int m = 1; m < 128; ++m) {           // pairs (i, i^m): disjoint involution
      int hb = 31 - __clz(m);
      if (tid < 64) {
        int low = tid & ((1 << hb) - 1);
        int p = ((tid >> hb) << (hb + 1)) | low;
        int q = p ^ m;
        float app = A[p * 128 + p], aqq = A[q * 128 + q], apq = A[p * 128 + q];
        float c = 1.0f, s = 0.0f;
        if (fabsf(apq) > 1e-12f) {
          float tau = (aqq - app) / (2.0f * apq);
          float t = ((tau >= 0.0f) ? 1.0f : -1.0f) / (fabsf(tau) + sqrtf(1.0f + tau * tau));
          c = rsqrtf(1.0f + t * t);
          s = t * c;
        }
        cs[2 * tid] = c; cs[2 * tid + 1] = s;
      }
      __syncthreads();
      for (int w = tid; w < 8192; w += 256) { // row rotations (Jt * A)
        int pr = w >> 7, j = w & 127;
        int low = pr & ((1 << hb) - 1);
        int p = ((pr >> hb) << (hb + 1)) | low;
        int q = p ^ m;
        float c = cs[2 * pr], s = cs[2 * pr + 1];
        float ap = A[p * 128 + j], aq = A[q * 128 + j];
        A[p * 128 + j] = c * ap - s * aq;
        A[q * 128 + j] = s * ap + c * aq;
      }
      __syncthreads();
      for (int w = tid; w < 8192; w += 256) { // column rotations (A * J) and V
        int pr = w >> 7, j = w & 127;
        int low = pr & ((1 << hb) - 1);
        int p = ((pr >> hb) << (hb + 1)) | low;
        int q = p ^ m;
        float c = cs[2 * pr], s = cs[2 * pr + 1];
        float ap = A[j * 128 + p], aq = A[j * 128 + q];
        A[j * 128 + p] = c * ap - s * aq;
        A[j * 128 + q] = s * ap + c * aq;
        float vp = V[j * 128 + p], vq = V[j * 128 + q];
        V[j * 128 + p] = c * vp - s * vq;
        V[j * 128 + q] = s * vp + c * vq;
      }
      __syncthreads();
    }
  }
  if (tid < 128) logl[tid] = __logf(fmaxf(A[tid * 128 + tid], 1e-6f));
  __syncthreads();
  // log_cov[i][j] = sum_k V[i][k] * logl[k] * V[j][k]; then row softmax -> adj
  int wave = tid >> 5, lane = tid & 31;
  for (int r = wave; r < 128; r += 8) {
    float acc0 = 0, acc1 = 0, acc2 = 0, acc3 = 0;
    for (int k = 0; k < 128; ++k) {
      float vr = V[r * 128 + k] * logl[k];
      acc0 += vr * V[lane * 128 + k];
      acc1 += vr * V[(lane + 32) * 128 + k];
      acc2 += vr * V[(lane + 64) * 128 + k];
      acc3 += vr * V[(lane + 96) * 128 + k];
    }
    float mx = fmaxf(fmaxf(acc0, acc1), fmaxf(acc2, acc3));
    for (int o = 16; o > 0; o >>= 1) mx = fmaxf(mx, __shfl_xor(mx, o, 32));
    float e0 = __expf(acc0 - mx), e1 = __expf(acc1 - mx);
    float e2 = __expf(acc2 - mx), e3 = __expf(acc3 - mx);
    float sm = e0 + e1 + e2 + e3;
    for (int o = 16; o > 0; o >>= 1) sm += __shfl_xor(sm, o, 32);
    float iv = 1.0f / sm;
    Aout[r * 128 + lane]      = e0 * iv;
    Aout[r * 128 + lane + 32] = e1 * iv;
    Aout[r * 128 + lane + 64] = e2 * iv;
    Aout[r * 128 + lane + 96] = e3 * iv;
  }
}

// --- K4: dilated 3-tap conv as im2col WMMA GEMM (+ optional fused residual GEMM)
__global__ void k_conv(const float* __restrict__ in, const float* __restrict__ w,
                       const float* __restrict__ bias, float* __restrict__ out,
                       const float* __restrict__ resw, const float* __restrict__ resb,
                       float* __restrict__ resout,
                       int Cin, int dil, int transposed) {
  extern __shared__ char dynsm[];
  _Float16* xs = (_Float16*)dynsm;
  const int b = blockIdx.x;
  const int t0 = blockIdx.y * 128;
  const int tid = threadIdx.x;
  const int K = 3 * Cin;
  const int rows = 128 + 2 * dil;
  _Float16* wt = xs + rows * Cin;      // weights transposed: wt[n*K + k]
  _Float16* rwt = wt + 64 * K;         // residual weights transposed: rwt[n*CN + k]
  for (int idx = tid; idx < rows * Cin; idx += 256) {
    int r = idx / Cin, c = idx - r * Cin;
    int t = t0 - dil + r;
    float v = 0.0f;
    if (t >= 0 && t < TN)
      v = transposed ? in[((size_t)b * CN + c) * TN + t]
                     : in[((size_t)b * TN + t) * FN + c];
    xs[r * Cin + c] = (_Float16)v;
  }
  for (int idx = tid; idx < 64 * K; idx += 256) {
    int n = idx / K, k = idx - n * K;
    wt[n * K + k] = (_Float16)w[k * FN + n];
  }
  if (resw) {
    for (int idx = tid; idx < 64 * CN; idx += 256) {
      int n = idx / CN, k = idx - n * CN;
      rwt[n * CN + k] = (_Float16)resw[k * FN + n];
    }
  }
  __syncthreads();
  const int wave = tid >> 5, lane = tid & 31;
  const int m0 = wave * 16;
  const int nch = K >> 5;
  v8f acc[4] = {{}, {}, {}, {}};
  for (int kc = 0; kc < nch; ++kc) {
    int k0 = kc << 5;
    int tap = k0 / Cin, kin = k0 - tap * Cin;
    v16h a = ld_a(xs + (m0 + tap * dil) * Cin + kin, Cin, lane);
#pragma unroll
    for (int nt = 0; nt < 4; ++nt) {
      v16h bb = ld_bt(wt + (nt * 16) * K + k0, K, lane);
      acc[nt] = WMMA_F16(a, bb, acc[nt]);
    }
  }
  const int n_lo = lane & 15;
  const int hi8 = ((lane >> 4) & 1) ? 8 : 0;
#pragma unroll
  for (int nt = 0; nt < 4; ++nt) {
    int n = nt * 16 + n_lo;
    float bv = bias[n];
#pragma unroll
    for (int r = 0; r < 8; ++r) {
      int t = t0 + m0 + hi8 + r;
      if (t < TN) out[((size_t)b * TN + t) * FN + n] = acc[nt][r] + bv;
    }
  }
  if (resw) {                                    // X_t @ res1_w + res1_b (center tap)
    v8f ar[4] = {{}, {}, {}, {}};
    for (int kc = 0; kc < (CN >> 5); ++kc) {
      int k0 = kc << 5;
      v16h a = ld_a(xs + (m0 + dil) * Cin + k0, Cin, lane);
#pragma unroll
      for (int nt = 0; nt < 4; ++nt) {
        v16h bb = ld_bt(rwt + (nt * 16) * CN + k0, CN, lane);
        ar[nt] = WMMA_F16(a, bb, ar[nt]);
      }
    }
#pragma unroll
    for (int nt = 0; nt < 4; ++nt) {
      int n = nt * 16 + n_lo;
      float bv = resb[n];
#pragma unroll
      for (int r = 0; r < 8; ++r) {
        int t = t0 + m0 + hi8 + r;
        if (t < TN) resout[((size_t)b * TN + t) * FN + n] = ar[nt][r] + bv;
      }
    }
  }
}

// ---- K5: y = relu(LN(raw)) + res (in-place into res buffer); opt f16-T copy -
__global__ void k_lnrr(const float* __restrict__ raw, const float* __restrict__ gam,
                       const float* __restrict__ bet, float* __restrict__ io,
                       _Float16* __restrict__ t16T, int write16) {
  int row = blockIdx.x;                 // b*TN + t
  int f = threadIdx.x;                  // 0..63
  float x = raw[(size_t)row * FN + f];
  __shared__ float r1[64], r2[64];
  r1[f] = x; r2[f] = x * x;
  __syncthreads();
  for (int o = 32; o > 0; o >>= 1) {
    if (f < o) { r1[f] += r1[f + o]; r2[f] += r2[f + o]; }
    __syncthreads();
  }
  float m = r1[0] * (1.0f / 64.0f);
  float var = r2[0] * (1.0f / 64.0f) - m * m;
  float y = (x - m) * rsqrtf(var + 1e-6f) * gam[f] + bet[f];
  y = fmaxf(y, 0.0f) + io[(size_t)row * FN + f];
  io[(size_t)row * FN + f] = y;
  if (write16) {
    int b = row / TN, t = row - b * TN;
    t16T[((size_t)b * FN + f) * TPAD + t] = (_Float16)y;
  }
}

__global__ void k_zpad16(_Float16* __restrict__ t16T) {
  int idx = blockIdx.x * 256 + threadIdx.x;   // BN*FN*(TPAD-TN)
  if (idx >= BN * FN * (TPAD - TN)) return;
  int fr = idx / (TPAD - TN);
  int t = TN + idx - fr * (TPAD - TN);
  t16T[(size_t)fr * TPAD + t] = (_Float16)0.0f;
}

// ---------------- K6: node_feats = X @ tcn_out  (WMMA over K=T) -------------
__global__ void k_nf(const _Float16* __restrict__ Xh, const _Float16* __restrict__ t16T,
                     float* __restrict__ nf) {
  int b = blockIdx.x;
  int wave = threadIdx.x >> 5, lane = threadIdx.x & 31;
  const _Float16* Xb = Xh + (size_t)b * CN * TPAD;
  const _Float16* Tb = t16T + (size_t)b * FN * TPAD;
  int m0 = wave * 16;
  v8f acc[4] = {{}, {}, {}, {}};
  for (int k0 = 0; k0 < TPAD; k0 += 32) {
    v16h a = ld_a(Xb + (size_t)m0 * TPAD + k0, TPAD, lane);
#pragma unroll
    for (int nt = 0; nt < 4; ++nt) {
      v16h bb = ld_bt(Tb + (size_t)(nt * 16) * TPAD + k0, TPAD, lane);
      acc[nt] = WMMA_F16(a, bb, acc[nt]);
    }
  }
  int n_lo = lane & 15;
  int hi8 = ((lane >> 4) & 1) ? 8 : 0;
#pragma unroll
  for (int nt = 0; nt < 4; ++nt)
#pragma unroll
    for (int r = 0; r < 8; ++r)
      nf[((size_t)b * CN + m0 + hi8 + r) * FN + nt * 16 + n_lo] = acc[nt][r];
}

// ---------------- K7: geo = relu(LN(adj @ node_feats)) ----------------------
__global__ void k_geo(const float* __restrict__ adj, const float* __restrict__ nf,
                      const float* __restrict__ gam, const float* __restrict__ bet,
                      float* __restrict__ geo) {
  __shared__ float gs[128 * 64];
  int b = blockIdx.x, tid = threadIdx.x;
  const float* Ab = adj + (size_t)b * 16384;
  const float* Nb = nf + (size_t)b * 8192;
  for (int idx = tid; idx < 8192; idx += 256) {
    int i = idx >> 6, j = idx & 63;
    float acc = 0.0f;
    for (int k = 0; k < 128; ++k) acc += Ab[i * 128 + k] * Nb[k * 64 + j];
    gs[idx] = acc;
  }
  __syncthreads();
  if (tid < 128) {
    float m = 0.0f, v = 0.0f;
    for (int j = 0; j < 64; ++j) { float x = gs[tid * 64 + j]; m += x; v += x * x; }
    m *= (1.0f / 64.0f);
    v = v * (1.0f / 64.0f) - m * m;
    float rs = rsqrtf(v + 1e-6f);
    for (int j = 0; j < 64; ++j) {
      float y = (gs[tid * 64 + j] - m) * rs * gam[j] + bet[j];
      geo[(size_t)b * 8192 + tid * 64 + j] = fmaxf(y, 0.0f);
    }
  }
}

// ---------------- K8: decay = sigmoid(gflat @ decay_w + decay_b) ------------
// 32KB row of gflat staged into LDS via the Tensor Data Mover when available.
__global__ void k_decay(const float* __restrict__ geo, const float* __restrict__ dw,
                        const float* __restrict__ db, float* __restrict__ dec) {
  __shared__ float g[8192];
  int b = blockIdx.x, h = threadIdx.x;   // 128 threads
#if defined(HAVE_TDM)
  if (h < 32) {
    const float* src = geo + (size_t)b * 8192;
    unsigned long long ga = (unsigned long long)(uintptr_t)src;
    unsigned int lds_lo = (unsigned int)(uintptr_t)(&g[0]);  // LDS offset = low 32 bits
    // D# group0: count=1, lds_addr, global_addr[56:0], type=2 ("image")
    v4u g0;
    g0[0] = 1u;
    g0[1] = lds_lo;
    g0[2] = (unsigned int)(ga & 0xFFFFFFFFu);
    g0[3] = ((unsigned int)((ga >> 32) & 0x01FFFFFFu)) | 0x80000000u;
    // D# group1: data_size=4B, tensor_dim0=8192, tensor_dim1=1, tile_dim0=8192,
    // tile_dim1=1, tensor_dim0_stride=8192
    v8i_t g1;
    g1[0] = 0x00020000;              // data_size=2 (4 bytes) at bits [17:16]
    g1[1] = (int)(8192u << 16);      // tensor_dim0[15:0] at bits [63:48]
    g1[2] = (int)(1u << 16);         // tensor_dim0[31:16]=0, tensor_dim1[15:0]=1
    g1[3] = (int)(8192u << 16);      // tensor_dim1[31:16]=0, tile_dim0=8192
    g1[4] = 1;                       // tile_dim1=1, tile_dim2=0
    g1[5] = 8192;                    // tensor_dim0_stride[31:0]
    g1[6] = 0;                       // stride hi, tensor_dim1_stride lo
    g1[7] = 0;
    v4i_t gz = {0, 0, 0, 0};
#if __clang_major__ >= 23
    v8i_t gz8 = {0, 0, 0, 0, 0, 0, 0, 0};
    __builtin_amdgcn_tensor_load_to_lds(g0, g1, gz, gz, gz8, 0);
#else
    __builtin_amdgcn_tensor_load_to_lds(g0, g1, gz, gz, 0);
#endif
    __builtin_amdgcn_s_wait_tensorcnt(0);
  }
  __syncthreads();
#else
  for (int i = h; i < 8192; i += 128) g[i] = geo[(size_t)b * 8192 + i];
  __syncthreads();
#endif
  float acc = db[h];
  for (int k = 0; k < 8192; ++k) acc += g[k] * dw[(size_t)k * HN + h];
  dec[b * HN + h] = 1.0f / (1.0f + __expf(-acc));
}

// ---------------- K9/K10: seq_in = mean_t(tcn_out) @ seq_w + seq_b ----------
__global__ void k_tavg(const float* __restrict__ tcn, float* __restrict__ tavg) {
  int bs = blockIdx.x;                 // b*SEQN + s
  int f = threadIdx.x;                 // 64
  int b = bs / SEQN, s = bs - b * SEQN;
  const float* p = tcn + (((size_t)b * TN) + (size_t)s * (TN / SEQN)) * FN + f;
  float acc = 0.0f;
  for (int i = 0; i < TN / SEQN; ++i) acc += p[(size_t)i * FN];
  tavg[(size_t)bs * FN + f] = acc * (1.0f / (float)(TN / SEQN));
}

__global__ void k_seqin(const float* __restrict__ tavg, const float* __restrict__ sw,
                        const float* __restrict__ sb, float* __restrict__ seqin) {
  int bs = blockIdx.x;
  int h = threadIdx.x;                 // 128
  float acc = sb[h];
  for (int f = 0; f < FN; ++f) acc += tavg[(size_t)bs * FN + f] * sw[f * HN + h];
  seqin[(size_t)bs * HN + h] = acc;
}

// ---------------- K11: LIF scan + attention + MLP head ----------------------
__global__ void k_init_out(float* __restrict__ o) { if (threadIdx.x == 0) o[512] = 0.0f; }

__global__ void k_head(const float* __restrict__ seqin, const float* __restrict__ dec,
                       const float* __restrict__ aw, const float* __restrict__ ab,
                       const float* __restrict__ fw, const float* __restrict__ fb,
                       const float* __restrict__ ow, const float* __restrict__ ob,
                       float* __restrict__ outp) {
  __shared__ float spk[SEQN * HN];
  __shared__ float wa[SEQN];
  __shared__ float feat[HN];
  __shared__ float f2[64];
  __shared__ float red[HN];
  int b = blockIdx.x, h = threadIdx.x;  // 128 threads
  float d = dec[b * HN + h];
  float mem = 0.0f;
  for (int t = 0; t < SEQN; ++t) {
    mem = mem * d + seqin[((size_t)b * SEQN + t) * HN + h];
    float s = (mem - 0.5f > 0.0f) ? 1.0f : 0.0f;
    mem -= s * 0.5f;
    spk[t * HN + h] = s;
  }
  __syncthreads();
  if (h < SEQN) {
    float a = ab[0];
    for (int k = 0; k < HN; ++k) a += spk[h * HN + k] * aw[k];
    wa[h] = a;
  }
  __syncthreads();
  if (h == 0) {
    float mx = wa[0];
    for (int t = 1; t < SEQN; ++t) mx = fmaxf(mx, wa[t]);
    float s = 0.0f;
    for (int t = 0; t < SEQN; ++t) { wa[t] = __expf(wa[t] - mx); s += wa[t]; }
    float iv = 1.0f / s;
    for (int t = 0; t < SEQN; ++t) wa[t] *= iv;
  }
  __syncthreads();
  float ft = 0.0f, ssum = 0.0f;
  for (int t = 0; t < SEQN; ++t) { float s = spk[t * HN + h]; ft += s * wa[t]; ssum += s; }
  feat[h] = ft;
  red[h] = ssum;
  __syncthreads();
  for (int o = 64; o > 0; o >>= 1) {
    if (h < o) red[h] += red[h + o];
    __syncthreads();
  }
  if (h == 0) atomicAdd(outp + 512, red[0] * (1.0f / ((float)BN * SEQN * HN)));
  if (h < 64) {
    float a = fb[h];
    for (int k = 0; k < HN; ++k) a += feat[k] * fw[k * 64 + h];
    f2[h] = fmaxf(a, 0.0f);
  }
  __syncthreads();
  if (h < 4) {
    float a = ob[h];
    for (int k = 0; k < 64; ++k) a += f2[k] * ow[k * 4 + h];
    outp[b * 4 + h] = a;
  }
}

// ---------------------------------------------------------------------------
extern "C" void kernel_launch(void* const* d_in, const int* in_sizes, int n_in,
                              void* d_out, int out_size, void* d_ws, size_t ws_size,
                              hipStream_t stream) {
  (void)in_sizes; (void)n_in; (void)out_size; (void)ws_size;
  const float* X       = (const float*)d_in[0];
  const float* conv1_w = (const float*)d_in[1];
  const float* conv1_b = (const float*)d_in[2];
  const float* ln1_s   = (const float*)d_in[3];
  const float* ln1_b   = (const float*)d_in[4];
  const float* res1_w  = (const float*)d_in[5];
  const float* res1_b  = (const float*)d_in[6];
  const float* conv2_w = (const float*)d_in[7];
  const float* conv2_b = (const float*)d_in[8];
  const float* ln2_s   = (const float*)d_in[9];
  const float* ln2_b   = (const float*)d_in[10];
  const float* conv3_w = (const float*)d_in[11];
  const float* conv3_b = (const float*)d_in[12];
  const float* ln3_s   = (const float*)d_in[13];
  const float* ln3_b   = (const float*)d_in[14];
  const float* lng_s   = (const float*)d_in[15];
  const float* lng_b   = (const float*)d_in[16];
  // d_in[17]/d_in[18]: mix_w/mix_b -> dead code in reference, skipped
  const float* seq_w   = (const float*)d_in[19];
  const float* seq_b   = (const float*)d_in[20];
  const float* decay_w = (const float*)d_in[21];
  const float* decay_b = (const float*)d_in[22];
  const float* attn_w  = (const float*)d_in[23];
  const float* attn_b  = (const float*)d_in[24];
  const float* feat_w  = (const float*)d_in[25];
  const float* feat_b  = (const float*)d_in[26];
  const float* out_w   = (const float*)d_in[27];
  const float* out_b   = (const float*)d_in[28];
  float* outp = (float*)d_out;

  char* ws = (char*)d_ws;
  size_t off = 0;
  auto alloc = [&](size_t bytes) -> void* {
    void* p = ws + off;
    off = (off + bytes + 255) & ~(size_t)255;
    return p;
  };
  _Float16* Xh    = (_Float16*)alloc((size_t)BN * CN * TPAD * 2);
  float*    mean  = (float*)alloc((size_t)BN * CN * 4);
  float*    cov   = (float*)alloc((size_t)BN * CN * CN * 4);
  float*    adj   = (float*)alloc((size_t)BN * CN * CN * 4);
  float*    bufA  = (float*)alloc((size_t)BN * TN * FN * 4);
  float*    bufB  = (float*)alloc((size_t)BN * TN * FN * 4);
  _Float16* t16T  = (_Float16*)alloc((size_t)BN * FN * TPAD * 2);
  float*    nf    = (float*)alloc((size_t)BN * CN * FN * 4);
  float*    geo   = (float*)alloc((size_t)BN * CN * FN * 4);
  float*    dec   = (float*)alloc((size_t)BN * HN * 4);
  float*    tavg  = (float*)alloc((size_t)BN * SEQN * FN * 4);
  float*    seqin = (float*)alloc((size_t)BN * SEQN * HN * 4);

  // 1. means + padded f16 X
  k_mean_xh<<<BN * CN, 256, 0, stream>>>(X, mean, Xh);
  // 2. covariance (WMMA, A-fragment reuse across 8 N-tiles)
  k_cov<<<BN, 256, 0, stream>>>(Xh, mean, cov);
  // 3. eigh -> log_cov -> softmax adj (LDS-resident Jacobi)
  k_eigh_adj<<<BN, 256, (size_t)(2 * 16384 + 256) * 4, stream>>>(cov, adj);
  // 4. TCN stack (WMMA convs + fused LN/ReLU/residual)
  dim3 cgrid(BN, (TN + 127) / 128);
  size_t sh1 = (size_t)((128 + 2) * CN + 64 * 3 * CN + 64 * CN) * 2;
  k_conv<<<cgrid, 256, sh1, stream>>>(X, conv1_w, conv1_b, bufB,
                                      res1_w, res1_b, bufA, CN, 1, 1);
  k_lnrr<<<BN * TN, 64, 0, stream>>>(bufB, ln1_s, ln1_b, bufA, nullptr, 0);   // bufA=t1
  size_t sh2 = (size_t)((128 + 4) * FN + 64 * 3 * FN) * 2;
  k_conv<<<cgrid, 256, sh2, stream>>>(bufA, conv2_w, conv2_b, bufB,
                                      nullptr, nullptr, nullptr, FN, 2, 0);
  k_lnrr<<<BN * TN, 64, 0, stream>>>(bufB, ln2_s, ln2_b, bufA, nullptr, 0);   // bufA=t2
  size_t sh3 = (size_t)((128 + 8) * FN + 64 * 3 * FN) * 2;
  k_conv<<<cgrid, 256, sh3, stream>>>(bufA, conv3_w, conv3_b, bufB,
                                      nullptr, nullptr, nullptr, FN, 4, 0);
  k_lnrr<<<BN * TN, 64, 0, stream>>>(bufB, ln3_s, ln3_b, bufA, t16T, 1);      // bufA=tcn_out
  k_zpad16<<<(BN * FN * (TPAD - TN) + 255) / 256, 256, 0, stream>>>(t16T);
  // 5. node_feats = X @ tcn_out (WMMA)
  k_nf<<<BN, 256, 0, stream>>>(Xh, t16T, nf);
  // 6. geo / gflat
  k_geo<<<BN, 256, 0, stream>>>(adj, nf, lng_s, lng_b, geo);
  // 7. decay gate (TDM-staged LDS when available)
  k_decay<<<BN, 128, 0, stream>>>(geo, decay_w, decay_b, dec);
  // 8. seq_in via time-averaged tcn_out
  k_tavg<<<BN * SEQN, 64, 0, stream>>>(bufA, tavg);
  k_seqin<<<BN * SEQN, 128, 0, stream>>>(tavg, seq_w, seq_b, seqin);
  // 9. LIF + attention + head
  k_init_out<<<1, 32, 0, stream>>>(outp);
  k_head<<<BN, 128, 0, stream>>>(seqin, dec, attn_w, attn_b, feat_w, feat_b,
                                 out_w, out_b, outp);
}
